// MS_SSA_Conv_4329327034383
// MI455X (gfx1250) — matrix operand order repaired and checked by reading, not compiled, not run on hardware
//
#include <hip/hip_runtime.h>
#include <hip/hip_bf16.h>

// MS_SSA_Conv (spiking self-attention) forward for gfx1250.
// GEMMs via v_wmma_f32_16x16x32_f16; activation panels staged to LDS with
// global_load_async_to_lds_b128; LIF recurrences fused into GEMM epilogues.

typedef _Float16 h8  __attribute__((ext_vector_type(8)));
typedef _Float16 h16 __attribute__((ext_vector_type(16)));
typedef float    vf8 __attribute__((ext_vector_type(8)));

static constexpr int TT = 4;      // time steps
static constexpr int BB = 32;     // batch
static constexpr int CC = 512;    // channels
static constexpr int NN = 256;    // H*W
static constexpr int HEADS = 8;
static constexpr int HD = 64;     // head dim
static constexpr int PS = 520;    // LDS panel row stride in halves (bank-safe)

__device__ __forceinline__ vf8 vzero8() {
  vf8 z;
#pragma unroll
  for (int i = 0; i < 8; ++i) z[i] = 0.0f;
  return z;
}

__device__ __forceinline__ h16 cat16(h8 lo, h8 hi) {
  h16 r;
#pragma unroll
  for (int i = 0; i < 8; ++i) { r[i] = lo[i]; r[i + 8] = hi[i]; }
  return r;
}

__device__ __forceinline__ h8 ldg8(const _Float16* p) { return *(const h8*)p; }

__device__ __forceinline__ vf8 wmma_f16(h16 a, h16 b, vf8 c) {
  return __builtin_amdgcn_wmma_f32_16x16x32_f16(false, a, false, b, (short)0, c,
                                                false, false);
}

// CDNA5 async global->LDS copy (ASYNCcnt-tracked, no VGPR staging).
__device__ __forceinline__ void async_ld16(unsigned lds_off, const void* gptr) {
  asm volatile("global_load_async_to_lds_b128 %0, %1, off"
               :: "v"(lds_off), "v"(gptr)
               : "memory");
}
__device__ __forceinline__ void wait_async0() {
  asm volatile("s_wait_asynccnt 0x0" ::: "memory");
}

// ---------------------------------------------------------------------------
// Fold BN scale into conv weight: Wf[d][c] = (f16)(W[d][c] * scale[d])
__global__ void fold_w_kernel(const float* __restrict__ W,
                              const float* __restrict__ scale,
                              _Float16* __restrict__ out) {
  int idx = blockIdx.x * blockDim.x + threadIdx.x;
  if (idx < CC * CC) {
    int d = idx >> 9;
    out[idx] = (_Float16)(W[idx] * scale[d]);
  }
}

// Proj bias fold: biasf = bp*p_scale + p_bias
__global__ void fold_pbias_kernel(const float* __restrict__ bp,
                                  const float* __restrict__ ps,
                                  const float* __restrict__ pb,
                                  float* __restrict__ out) {
  int i = blockIdx.x * blockDim.x + threadIdx.x;
  if (i < CC) out[i] = bp[i] * ps[i] + pb[i];
}

// ---------------------------------------------------------------------------
// shortcut LIF over x: spikes written transposed as xs_t[t][b][n][c] (f16)
__global__ void lif_input_kernel(const float* __restrict__ x,
                                 _Float16* __restrict__ xs_t) {
  int idx = blockIdx.x * blockDim.x + threadIdx.x;  // (b, c, n) n-fastest
  if (idx >= BB * CC * NN) return;
  const int n = idx & (NN - 1);
  const int c = (idx >> 8) & (CC - 1);
  const int b = idx >> 17;
  float v = 0.0f;
#pragma unroll
  for (int t = 0; t < TT; ++t) {
    float xv = x[(((size_t)(t * BB + b)) * CC + c) * NN + n];
    v = 0.5f * (v + xv);               // v += (x - v)/tau, tau = 2
    float s = (v >= 1.0f) ? 1.0f : 0.0f;
    v *= (1.0f - s);                   // hard reset
    xs_t[(((size_t)(t * BB + b)) * NN + n) * CC + c] = (_Float16)s;
  }
}

// ---------------------------------------------------------------------------
// QKV branch: y = (Wf @ xs) + bias per (t,b); LIF over t fused (state in VGPRs).
// MODE 0 (Q): out_sp  = [t][b][h][n][e] f16
// MODE 1 (K): out_sp  = [t][b][d][n]    f16
// MODE 2 (V): out_sp  = [t][b][d][n]    f16, out_f32 = [t][b][h][n][e] f32
template <int MODE>
__global__ __launch_bounds__(128) void qkv_gemm_lif_kernel(
    const _Float16* __restrict__ xs_t, const _Float16* __restrict__ Wf,
    const float* __restrict__ bias, _Float16* __restrict__ out_sp,
    float* __restrict__ out_f32) {
  extern __shared__ _Float16 panel[];  // [64 rows][PS halves]
  const int lane = threadIdx.x & 31;
  const int wave = threadIdx.x >> 5;
  const int b = blockIdx.x;
  const int d0 = blockIdx.y * 64 + wave * 16;
  const int n0 = blockIdx.z * 64;
  const int m = lane & 15;
  const int g = lane >> 4;
  const unsigned lds_base = (unsigned)(size_t)(void*)panel;

  vf8 st[4];
#pragma unroll
  for (int i = 0; i < 4; ++i) st[i] = vzero8();

  const _Float16* wrow = Wf + (size_t)(d0 + m) * CC;

  for (int t = 0; t < TT; ++t) {
    // ---- async-stage the 64x512 f16 activation panel (contiguous in global)
    const char* src =
        (const char*)(xs_t + (size_t)(t * BB + b) * NN * CC + (size_t)n0 * CC);
    for (int i = threadIdx.x; i < 4096; i += 128) {
      const int r = i >> 6, cchunk = i & 63;
      async_ld16(lds_base + (unsigned)(r * PS * 2 + cchunk * 16),
                 src + r * 1024 + cchunk * 16);
    }
    wait_async0();
    __syncthreads();

    vf8 acc[4];
#pragma unroll
    for (int i = 0; i < 4; ++i) acc[i] = vzero8();

    // ---- K loop: A (weights) double-buffered from global, B from LDS
    h16 A = cat16(ldg8(wrow + 8 * g), ldg8(wrow + 16 + 8 * g));
#pragma unroll
    for (int kk = 0; kk < 16; ++kk) {
      const int c0 = kk * 32;
      h16 An = A;
      if (kk < 15) {
        const int c1 = c0 + 32;
        An = cat16(ldg8(wrow + c1 + 8 * g), ldg8(wrow + c1 + 16 + 8 * g));
      }
      h16 Bv[4];
#pragma unroll
      for (int nt = 0; nt < 4; ++nt) {
        const _Float16* bp2 = &panel[(nt * 16 + m) * PS + c0 + 8 * g];
        Bv[nt] = cat16(*(const h8*)bp2, *(const h8*)(bp2 + 16));
      }
#pragma unroll
      for (int nt = 0; nt < 4; ++nt) acc[nt] = wmma_f16(A, Bv[nt], acc[nt]);
      A = An;
    }
    __syncthreads();  // all waves done with panel before next t overwrites it

    // ---- BN bias + LIF + spike emission
    const size_t tb = (size_t)(t * BB + b);
#pragma unroll
    for (int nt = 0; nt < 4; ++nt) {
#pragma unroll
      for (int j = 0; j < 8; ++j) {
        const int d = d0 + j + 8 * g;
        const int n = n0 + nt * 16 + m;
        float y = acc[nt][j] + bias[d];
        float v = 0.5f * (st[nt][j] + y);
        float s = (v >= 1.0f) ? 1.0f : 0.0f;
        st[nt][j] = v * (1.0f - s);
        const int head = d >> 6, e = d & 63;
        const size_t qi = ((tb * HEADS + head) * NN + n) * HD + e;
        if constexpr (MODE == 0) {
          out_sp[qi] = (_Float16)s;
        } else {
          out_sp[(tb * CC + d) * NN + n] = (_Float16)s;
          if constexpr (MODE == 2) out_f32[qi] = s;
        }
      }
    }
  }
}

// ---------------------------------------------------------------------------
// STAtten: per (b,head): attn = (1/16) k^T v per chunk; out = q@attn;
// attn-LIF (v_th=0.5) over t with membrane in LDS; emit x2 transposed
// x2_t[t][b][n][c] f16 spikes for the proj GEMM.
__global__ __launch_bounds__(128) void attention_kernel(
    const _Float16* __restrict__ q16, const _Float16* __restrict__ k16,
    const _Float16* __restrict__ v16, _Float16* __restrict__ x2t) {
  extern __shared__ char smem[];
  _Float16* st = (_Float16*)smem;                       // [256][64] membrane
  _Float16* attnT = (_Float16*)(smem + NN * HD * 2);    // [64][72] attn^T, padded

  const int lane = threadIdx.x & 31;
  const int wave = threadIdx.x >> 5;
  const int b = blockIdx.x;
  const int head = blockIdx.y;
  const int m = lane & 15;
  const int g = lane >> 4;

  for (int i = threadIdx.x; i < NN * HD; i += 128) st[i] = (_Float16)0.0f;
  __syncthreads();

  for (int ci = 0; ci < 2; ++ci) {
    // ---- attn(d,e) = sum_s k[s,d] v[s,e] ; wave owns d rows [16w,16w+16)
    vf8 aacc[4];
#pragma unroll
    for (int i = 0; i < 4; ++i) aacc[i] = vzero8();
    const int dw = wave * 16;
#pragma unroll 1
    for (int ct = 0; ct < 2; ++ct) {
      const int t = 2 * ci + ct;
      const _Float16* kb = k16 + ((size_t)(t * BB + b) * CC + head * HD) * NN;
      const _Float16* vb = v16 + ((size_t)(t * BB + b) * CC + head * HD) * NN;
      for (int ksn = 0; ksn < 8; ++ksn) {
        const int s0 = ksn * 32;
        const _Float16* ap = kb + (size_t)(dw + m) * NN + s0;
        h16 A = cat16(ldg8(ap + 8 * g), ldg8(ap + 16 + 8 * g));
#pragma unroll
        for (int et = 0; et < 4; ++et) {
          const _Float16* bp = vb + (size_t)(et * 16 + m) * NN + s0 + 8 * g;
          h16 Bv = cat16(ldg8(bp), ldg8(bp + 16));
          aacc[et] = wmma_f16(A, Bv, aacc[et]);
        }
      }
    }
    // transpose+scale into LDS: attnT[e][d]
#pragma unroll
    for (int et = 0; et < 4; ++et)
#pragma unroll
      for (int j = 0; j < 8; ++j) {
        const int d = dw + j + 8 * g;
        const int e = et * 16 + m;
        attnT[e * 72 + d] = (_Float16)(aacc[et][j] * 0.0625f);
      }
    __syncthreads();

    // ---- out[s,e] = q[s,:] @ attn ; LIF over t; wave owns n in [64w,64w+64)
#pragma unroll 1
    for (int ct = 0; ct < 2; ++ct) {
      const int t = 2 * ci + ct;
      const _Float16* qb = q16 + ((size_t)(t * BB + b) * HEADS + head) * NN * HD;
      _Float16* xo = x2t + (size_t)(t * BB + b) * NN * CC + head * HD;
#pragma unroll 1
      for (int mt = 0; mt < 4; ++mt) {
        const int n0 = wave * 64 + mt * 16;
#pragma unroll 1
        for (int et = 0; et < 4; ++et) {
          vf8 oacc = vzero8();
#pragma unroll
          for (int ksd = 0; ksd < 2; ++ksd) {
            const int dk = ksd * 32;
            const _Float16* ap = qb + (size_t)(n0 + m) * HD + dk;
            h16 A = cat16(ldg8(ap + 8 * g), ldg8(ap + 16 + 8 * g));
            const _Float16* bp = attnT + (et * 16 + m) * 72 + dk + 8 * g;
            h16 Bv = cat16(*(const h8*)bp, *(const h8*)(bp + 16));
            oacc = wmma_f16(A, Bv, oacc);
          }
#pragma unroll
          for (int j = 0; j < 8; ++j) {
            const int n = n0 + j + 8 * g;
            const int e = et * 16 + m;
            float vprev = (float)st[n * HD + e];
            float v = 0.5f * (vprev + oacc[j]);
            float s = (v >= 0.5f) ? 1.0f : 0.0f;
            st[n * HD + e] = (_Float16)(v * (1.0f - s));
            xo[(size_t)n * CC + e] = (_Float16)s;
          }
        }
      }
    }
    __syncthreads();  // before next chunk overwrites attnT
  }
}

// ---------------------------------------------------------------------------
// proj conv + folded BN bias + residual: out[t,b,d,n] (f32)
__global__ __launch_bounds__(128) void proj_residual_kernel(
    const _Float16* __restrict__ x2t, const _Float16* __restrict__ Wpf,
    const float* __restrict__ biasf, const float* __restrict__ xin,
    float* __restrict__ out) {
  extern __shared__ _Float16 panel[];  // [64][PS]
  const int lane = threadIdx.x & 31;
  const int wave = threadIdx.x >> 5;
  const int tb = blockIdx.x;  // t*BB + b
  const int d0 = blockIdx.y * 64 + wave * 16;
  const int n0 = blockIdx.z * 64;
  const int m = lane & 15;
  const int g = lane >> 4;
  const unsigned lds_base = (unsigned)(size_t)(void*)panel;

  const char* src = (const char*)(x2t + (size_t)tb * NN * CC + (size_t)n0 * CC);
  for (int i = threadIdx.x; i < 4096; i += 128) {
    const int r = i >> 6, cchunk = i & 63;
    async_ld16(lds_base + (unsigned)(r * PS * 2 + cchunk * 16),
               src + r * 1024 + cchunk * 16);
  }
  wait_async0();
  __syncthreads();

  vf8 acc[4];
#pragma unroll
  for (int i = 0; i < 4; ++i) acc[i] = vzero8();

  const _Float16* wrow = Wpf + (size_t)(d0 + m) * CC;
  h16 A = cat16(ldg8(wrow + 8 * g), ldg8(wrow + 16 + 8 * g));
#pragma unroll
  for (int kk = 0; kk < 16; ++kk) {
    const int c0 = kk * 32;
    h16 An = A;
    if (kk < 15) {
      const int c1 = c0 + 32;
      An = cat16(ldg8(wrow + c1 + 8 * g), ldg8(wrow + c1 + 16 + 8 * g));
    }
    h16 Bv[4];
#pragma unroll
    for (int nt = 0; nt < 4; ++nt) {
      const _Float16* bp2 = &panel[(nt * 16 + m) * PS + c0 + 8 * g];
      Bv[nt] = cat16(*(const h8*)bp2, *(const h8*)(bp2 + 16));
    }
#pragma unroll
    for (int nt = 0; nt < 4; ++nt) acc[nt] = wmma_f16(A, Bv[nt], acc[nt]);
    A = An;
  }

#pragma unroll
  for (int nt = 0; nt < 4; ++nt)
#pragma unroll
    for (int j = 0; j < 8; ++j) {
      const int d = d0 + j + 8 * g;
      const int n = n0 + nt * 16 + m;
      const size_t idx = ((size_t)tb * CC + d) * NN + n;
      out[idx] = acc[nt][j] + biasf[d] + xin[idx];
    }
}

// ---------------------------------------------------------------------------
extern "C" void kernel_launch(void* const* d_in, const int* in_sizes, int n_in,
                              void* d_out, int out_size, void* d_ws,
                              size_t ws_size, hipStream_t stream) {
  const float* x = (const float*)d_in[0];
  const float* Wq = (const float*)d_in[1];
  const float* q_scale = (const float*)d_in[2];
  const float* q_bias = (const float*)d_in[3];
  const float* Wk = (const float*)d_in[4];
  const float* k_scale = (const float*)d_in[5];
  const float* k_bias = (const float*)d_in[6];
  const float* Wv = (const float*)d_in[7];
  const float* v_scale = (const float*)d_in[8];
  const float* v_bias = (const float*)d_in[9];
  const float* Wp = (const float*)d_in[10];
  const float* bp = (const float*)d_in[11];
  const float* p_scale = (const float*)d_in[12];
  const float* p_bias = (const float*)d_in[13];

  const size_t ACT = (size_t)TT * BB * CC * NN;  // 16.7M elements
  char* ws = (char*)d_ws;
  size_t off = 0;
  auto take = [&](size_t bytes) {
    char* p = ws + off;
    off = (off + bytes + 255) & ~(size_t)255;
    return p;
  };
  _Float16* xs_t = (_Float16*)take(ACT * 2);
  _Float16* q16 = (_Float16*)take(ACT * 2);
  _Float16* k16 = (_Float16*)take(ACT * 2);
  _Float16* v16 = (_Float16*)take(ACT * 2);
  _Float16* x2t = (_Float16*)take(ACT * 2);
  _Float16* Wqf = (_Float16*)take((size_t)CC * CC * 2);
  _Float16* Wkf = (_Float16*)take((size_t)CC * CC * 2);
  _Float16* Wvf = (_Float16*)take((size_t)CC * CC * 2);
  _Float16* Wpf = (_Float16*)take((size_t)CC * CC * 2);
  float* pbiasf = (float*)take(CC * 4);

  float* out1 = (float*)d_out;         // x3 + identity
  float* outv = out1 + ACT;            // v spikes (f32)

  // weight folds
  fold_w_kernel<<<(CC * CC + 255) / 256, 256, 0, stream>>>(Wq, q_scale, Wqf);
  fold_w_kernel<<<(CC * CC + 255) / 256, 256, 0, stream>>>(Wk, k_scale, Wkf);
  fold_w_kernel<<<(CC * CC + 255) / 256, 256, 0, stream>>>(Wv, v_scale, Wvf);
  fold_w_kernel<<<(CC * CC + 255) / 256, 256, 0, stream>>>(Wp, p_scale, Wpf);
  fold_pbias_kernel<<<2, 256, 0, stream>>>(bp, p_scale, p_bias, pbiasf);

  // shortcut LIF + transpose
  lif_input_kernel<<<(BB * CC * NN) / 256, 256, 0, stream>>>(x, xs_t);

  // QKV fused GEMM+BN+LIF (dynamic LDS: 64 x PS halves panel)
  const int panel_bytes = 64 * PS * 2;  // 66560
  dim3 ggrid(BB, CC / 64, NN / 64);
  qkv_gemm_lif_kernel<0><<<ggrid, 128, panel_bytes, stream>>>(xs_t, Wqf, q_bias,
                                                              q16, nullptr);
  qkv_gemm_lif_kernel<1><<<ggrid, 128, panel_bytes, stream>>>(xs_t, Wkf, k_bias,
                                                              k16, nullptr);
  qkv_gemm_lif_kernel<2><<<ggrid, 128, panel_bytes, stream>>>(xs_t, Wvf, v_bias,
                                                              v16, outv);

  // chunked linear attention + attn LIF
  const int smem_bytes = NN * HD * 2 + 64 * 72 * 2;  // 41984 B
  attention_kernel<<<dim3(BB, HEADS), 128, smem_bytes, stream>>>(q16, k16, v16,
                                                                 x2t);

  // proj + residual
  proj_residual_kernel<<<dim3(TT * BB, CC / 64, NN / 64), 128, panel_bytes,
                         stream>>>(x2t, Wpf, pbiasf, x, out1);

  (void)in_sizes; (void)n_in; (void)out_size; (void)ws_size;
}
